// RegionProposalNetwork_81114752352717
// MI455X (gfx1250) — matrix operand order repaired and checked by reading, not compiled, not run on hardware
//
#include <hip/hip_runtime.h>
#include <hip/hip_bf16.h>
#include <stdint.h>

// ---------------------------------------------------------------------------
// RPN forward for MI455X (gfx1250, wave32, WMMA + async global->LDS staging).
// 3x3 conv [32,2048,25,25] -> [32,9,25,25] as implicit GEMM
//   M=9(pad16) x N=20000 x K=18432 via v_wmma_f32_16x16x32_bf16.
// Input slabs are staged to LDS asynchronously (ASYNCcnt, double buffered) so
// the 9-tap / 8-wave reuse comes from LDS instead of 9x re-reads of L2/HBM:
// global traffic ~230MB (~10us floor at 23.3TB/s) instead of ~1.5GB.
// K is split 4-ways into deterministic partial buffers for occupancy
// (640 WGs / 5120 waves instead of 157 WGs).
// ---------------------------------------------------------------------------

typedef __attribute__((ext_vector_type(16))) __bf16 v16bf;
typedef __attribute__((ext_vector_type(8)))  float  v8f;

#define C_IN      2048
#define HID       9
#define BATCH     32
#define HW        25
#define NPOS_TOT  (BATCH * HW * HW)      // 20000
#define N_ANC     9
#define N_FLAT    (NPOS_TOT * N_ANC)     // 180000
#define N_SAMP    2048
#define KSPLIT    4
#define CPW       16                     // ci-chunks (of 32) per workgroup
#define CIW       225                    // LDS words per channel (8*28 + 1 pad)
#define XT_WORDS  (32 * CIW)             // words per stage buffer
#define STAGE_PER_THREAD 22              // async b32 loads per thread per stage

// ---------------- gfx1250 async global->LDS + asynccnt helpers -------------
typedef __attribute__((address_space(1))) int g_i32;
typedef __attribute__((address_space(3))) int l_i32;

__device__ __forceinline__ void async_copy_b32(const float* gp, float* lp) {
#if __has_builtin(__builtin_amdgcn_global_load_async_to_lds_b32)
    __builtin_amdgcn_global_load_async_to_lds_b32(
        (g_i32*)(uintptr_t)gp,
        (l_i32*)(uint32_t)(uintptr_t)lp,
        0, 0);
#else
    uint32_t l = (uint32_t)(uintptr_t)lp;
    asm volatile("global_load_async_to_lds_b32 %0, %1, off"
                 :: "v"(l), "v"(gp) : "memory");
#endif
}

__device__ __forceinline__ void wait_async_le(int n_const) {
#if __has_builtin(__builtin_amdgcn_s_wait_asynccnt)
    if (n_const == 0) __builtin_amdgcn_s_wait_asynccnt(0);
    else              __builtin_amdgcn_s_wait_asynccnt(STAGE_PER_THREAD);
#else
    if (n_const == 0) asm volatile("s_wait_asynccnt 0" ::: "memory");
    else              asm volatile("s_wait_asynccnt 22" ::: "memory");
#endif
}

// ---------------------------------------------------------------------------
// Prep: pack Wh (f32, OIHW [9,2048,3,3]) into bf16 A-fragments in the CDNA5
// 16-bit A(16x32) lane layout: lane l (M=l%16), elem i ->
//   K = (i<8 ? i : i+8) + (l>=16 ? 8 : 0);  k-order = khw*2048 + ci.
// ---------------------------------------------------------------------------
__global__ void pack_weights_kernel(const float* __restrict__ Wh,
                                    __bf16* __restrict__ packed)
{
    int q    = blockIdx.x;        // chunk 0..575  (khw*64 + cic)
    int tid  = threadIdx.x;       // 0..511
    int lane = tid >> 4;
    int i    = tid & 15;
    int khw  = q / 64;
    int cic  = q % 64;
    int m    = lane & 15;
    int koff = ((i < 8) ? i : i + 8) + ((lane >= 16) ? 8 : 0);
    int ci   = cic * 32 + koff;
    float w  = (m < HID) ? Wh[m * (C_IN * 9) + ci * 9 + khw] : 0.0f;
    packed[(size_t)q * 512 + tid] = (__bf16)w;
}

// ---------------------------------------------------------------------------
// Conv-hidden kernel.  grid = (KSPLIT, 5 row-blocks, 32 images), 256 threads.
// Each WG: rows r0..r0+4 of one image, 16 ci-chunks of its K split.
// Per chunk: async-stage [32ci x 7rows x 25cols] f32 slab to LDS (double
// buffered), then 8 waves x 9 taps of WMMA read it back as bf16 B-fragments.
// ---------------------------------------------------------------------------
__global__ __launch_bounds__(256)
void conv_hidden_kernel(const float*  __restrict__ X,
                        const __bf16* __restrict__ Wp,
                        float* __restrict__ part)   // [KSPLIT][20000][16]
{
    __shared__ float xs[2 * XT_WORDS];

    const int tid = threadIdx.x;
    const int ks  = blockIdx.x;          // K split 0..3
    const int rb  = blockIdx.y;          // row block 0..4
    const int b   = blockIdx.z;          // image 0..31
    const int r0  = rb * 5;

    // Zero both buffers once: col borders (cx=0,26,27) and invalid-row slots
    // stay zero forever (async stages never touch them).
    for (int i = tid; i < 2 * XT_WORDS; i += 256) xs[i] = 0.0f;
    __syncthreads();

    const int wid  = tid >> 5;
    const int lane = tid & 31;
    const int n    = lane & 15;
    const int hi8  = (lane >= 16) ? 8 : 0;
    const int hi16 = hi8 * 2;
    const int q    = wid * 16 + n;              // 0..127 (125 valid positions)
    const int qc   = (q < 125) ? q : 0;
    const int prow = qc / 25;
    const int pcol = qc - prow * 25;

    const float* Xb = X + (size_t)b * (C_IN * 625);

    // Issue one stage: 32ci x 7row-slots x 25cols = 5600 b32 (pad to 5632),
    // uniformly 22 async instructions per thread (ASYNCcnt bookkeeping).
    auto stage = [&](int cic_local, int buf) {
        const int cibase = (ks * CPW + cic_local) * 32;
        #pragma unroll
        for (int k = 0; k < STAGE_PER_THREAD; ++k) {
            const int e    = tid + k * 256;
            const int ci   = e / 175;            // 7*25
            const int rem  = e - ci * 175;
            const int rs   = rem / 25;           // row slot 0..6
            const int col  = rem - rs * 25;
            const int grow = r0 - 1 + rs;        // input row
            const bool ok  = (ci < 32) && (grow >= 0) && (grow < HW);
            const int gidx = (cibase + (ok ? ci : 0)) * 625 +
                             (ok ? grow : 0) * 25 + col;
            const int lidx = ok ? (buf * XT_WORDS + ci * CIW + rs * 28 + col + 1)
                                : (buf * XT_WORDS + 7 * 28);   // trash word
            async_copy_b32(Xb + gidx, &xs[lidx]);
        }
    };

    v8f acc = {};
    stage(0, 0);

    #pragma unroll 1
    for (int j = 0; j < CPW; ++j) {
        const int cur = j & 1;
        if (j + 1 < CPW) stage(j + 1, cur ^ 1);
        if (j + 1 < CPW) wait_async_le(STAGE_PER_THREAD);   // stage j landed
        else             wait_async_le(0);
        __syncthreads();                                    // publish buffer

        const float* xb = &xs[cur * XT_WORDS + hi8 * CIW + prow * 28 + pcol];
        const __bf16* apj = Wp + ((size_t)(ks * CPW + j)) * 512 + lane * 16;

        #pragma unroll
        for (int kh = 0; kh < 3; ++kh) {
            #pragma unroll
            for (int kw = 0; kw < 3; ++kw) {
                const int khw = kh * 3 + kw;
                // A fragment: one 32B vector load (pre-packed bf16 layout).
                v16bf a = *(const v16bf*)(apj + (size_t)khw * (64 * 512));
                // B fragment from LDS: 16-bit B(32x16) layout, K = (i<8 ? i : i+8) + hi8.
                const float* bp = xb + kh * 28 + kw;
                v16bf bfr;
                #pragma unroll
                for (int i = 0; i < 16; ++i)
                    bfr[i] = (__bf16)bp[((i < 8) ? i : i + 8) * CIW];
                acc = __builtin_amdgcn_wmma_f32_16x16x32_bf16(
                    false, a, false, bfr, (short)0, acc, false, false);
            }
        }
        __syncthreads();   // before next stage overwrites buffer `cur`
    }

    if (q < 125) {
        const int p = b * 625 + (r0 + prow) * 25 + pcol;
        float* pp = part + ((size_t)ks * NPOS_TOT + p) * 16;
        #pragma unroll
        for (int r = 0; r < 8; ++r) pp[r + hi8] = acc[r];
    }
}

// ---------------------------------------------------------------------------
// Epilogue: deterministic K-split reduction + bias + ReLU + both 1x1 convs.
// ---------------------------------------------------------------------------
__global__ __launch_bounds__(256)
void finish_kernel(const float* __restrict__ part,
                   const float* __restrict__ bh,
                   const float* __restrict__ Wc, const float* __restrict__ bc,
                   const float* __restrict__ Wr, const float* __restrict__ br,
                   float* __restrict__ conf_ws, float* __restrict__ off_ws)
{
    __shared__ float sW[45 * 9];
    __shared__ float sB[45];
    __shared__ float sbh[9];

    const int tid = threadIdx.x;
    if (tid < 45 * 9) sW[tid] = (tid < 81) ? Wc[tid] : Wr[tid - 81];
    if (tid < 45)     sB[tid] = (tid < 9)  ? bc[tid] : br[tid - 9];
    if (tid < 9)      sbh[tid] = bh[tid];
    __syncthreads();

    const int p = blockIdx.x * 256 + tid;
    if (p >= NPOS_TOT) return;

    float y[HID];
    #pragma unroll
    for (int m = 0; m < HID; ++m) y[m] = sbh[m];
    #pragma unroll
    for (int ks = 0; ks < KSPLIT; ++ks) {
        const float* pp = part + ((size_t)ks * NPOS_TOT + p) * 16;
        #pragma unroll
        for (int m = 0; m < HID; ++m) y[m] += pp[m];
    }
    #pragma unroll
    for (int m = 0; m < HID; ++m) y[m] = fmaxf(y[m], 0.0f);

    for (int a = 0; a < 45; ++a) {
        float s = sB[a];
        #pragma unroll
        for (int c = 0; c < HID; ++c) s += sW[a * 9 + c] * y[c];
        if (a < 9) conf_ws[(size_t)p * 9  + a]       = s;
        else       off_ws [(size_t)p * 36 + (a - 9)] = s;
    }
}

// ---------------------------------------------------------------------------
// Gather + proposal generation.  off_ws flat [p*36+c] == off_flat[idx*4+d].
// ---------------------------------------------------------------------------
__global__ void gather_kernel(const float* __restrict__ conf_ws,
                              const float* __restrict__ off_ws,
                              const int*   __restrict__ pos_idx,
                              const int*   __restrict__ neg_idx,
                              const float* __restrict__ pos_ancs,
                              float* __restrict__ out)
{
    const int i = blockIdx.x * blockDim.x + threadIdx.x;
    if (i >= N_SAMP) return;

    float* pos_conf = out;
    float* neg_conf = out + N_SAMP;
    float* pos_off  = out + 2 * N_SAMP;
    float* props    = out + 6 * N_SAMP;

    const int pi = pos_idx[i];
    const int ni = neg_idx[i];
    pos_conf[i] = conf_ws[pi];
    neg_conf[i] = conf_ws[ni];

    const float o0 = off_ws[(size_t)pi * 4 + 0];
    const float o1 = off_ws[(size_t)pi * 4 + 1];
    const float o2 = off_ws[(size_t)pi * 4 + 2];
    const float o3 = off_ws[(size_t)pi * 4 + 3];
    pos_off[i * 4 + 0] = o0;
    pos_off[i * 4 + 1] = o1;
    pos_off[i * 4 + 2] = o2;
    pos_off[i * 4 + 3] = o3;

    const float x1 = pos_ancs[i * 4 + 0], y1 = pos_ancs[i * 4 + 1];
    const float x2 = pos_ancs[i * 4 + 2], y2 = pos_ancs[i * 4 + 3];
    const float cx = (x1 + x2) * 0.5f, cy = (y1 + y2) * 0.5f;
    const float w  = x2 - x1,          h  = y2 - y1;
    const float ncx = cx + o0 * w;
    const float ncy = cy + o1 * h;
    const float nw  = w * expf(o2);
    const float nh  = h * expf(o3);
    props[i * 4 + 0] = ncx - nw * 0.5f;
    props[i * 4 + 1] = ncy - nh * 0.5f;
    props[i * 4 + 2] = ncx + nw * 0.5f;
    props[i * 4 + 3] = ncy + nh * 0.5f;
}

// ---------------------------------------------------------------------------
extern "C" void kernel_launch(void* const* d_in, const int* in_sizes, int n_in,
                              void* d_out, int out_size, void* d_ws, size_t ws_size,
                              hipStream_t stream)
{
    const float* X        = (const float*)d_in[0];
    const int*   pos_idx  = (const int*)  d_in[1];
    const int*   neg_idx  = (const int*)  d_in[2];
    const float* pos_ancs = (const float*)d_in[3];
    const float* Wh       = (const float*)d_in[4];
    const float* bh       = (const float*)d_in[5];
    const float* Wc       = (const float*)d_in[6];
    const float* bc       = (const float*)d_in[7];
    const float* Wr       = (const float*)d_in[8];
    const float* br       = (const float*)d_in[9];

    // ws: conf_flat | off_flat | packed bf16 weights | K-split partials
    float*  conf_ws = (float*)d_ws;
    float*  off_ws  = conf_ws + N_FLAT;
    __bf16* Wp      = (__bf16*)(off_ws + (size_t)NPOS_TOT * 36);
    float*  part    = (float*)(Wp + (size_t)576 * 512);

    pack_weights_kernel<<<576, 512, 0, stream>>>(Wh, Wp);
    conv_hidden_kernel<<<dim3(KSPLIT, 5, BATCH), 256, 0, stream>>>(X, Wp, part);
    finish_kernel<<<(NPOS_TOT + 255) / 256, 256, 0, stream>>>(
        part, bh, Wc, bc, Wr, br, conf_ws, off_ws);
    gather_kernel<<<(N_SAMP + 255) / 256, 256, 0, stream>>>(
        conf_ws, off_ws, pos_idx, neg_idx, pos_ancs, (float*)d_out);
}
